// FeedForwardNetwork_57793079935381
// MI455X (gfx1250) — matrix operand order repaired and checked by reading, not compiled, not run on hardware
//
#include <hip/hip_runtime.h>

// ---------------- types / helpers ----------------
typedef __attribute__((ext_vector_type(16))) __bf16 v16bf;
typedef __attribute__((ext_vector_type(8)))  float  v8f;
typedef __attribute__((ext_vector_type(4)))  int    v4i;

union Frag { v16bf v; unsigned int u[8]; uint4 q[2]; };

__device__ __forceinline__ unsigned short f2bf(float f) {
  unsigned int x = __float_as_uint(f);
  x += 0x7FFFu + ((x >> 16) & 1u);          // round-to-nearest-even
  return (unsigned short)(x >> 16);
}
// packed pair conversion: hardware v_cvt_pk_bf16_f32 when available
__device__ __forceinline__ unsigned int cvt2bf(float a, float b) {
#if __has_builtin(__builtin_amdgcn_cvt_pk_bf16_f32)
  auto r = __builtin_amdgcn_cvt_pk_bf16_f32(a, b);
  unsigned int u; __builtin_memcpy(&u, &r, sizeof(u));
  return u;
#else
  return (unsigned int)f2bf(a) | ((unsigned int)f2bf(b) << 16);
#endif
}

// ---------------- problem constants ----------------
#define HDIM 512      // hidden
#define FDIM 2048     // filter
#define TM   64       // tokens per WG
#define FC   64       // F-chunk width
#define NFC  (FDIM / FC)   // 32 chunks
#define KP   128      // K panel for phase A
#define W1NCG 4       // col-groups in W1 panel (64 cols)
#define W2NCG 32      // col-groups in W2 chunk (512 cols)

// B layout (compact, conflict-free frag b128 loads, sequential staging):
//   word(ks,cg,part,L,v4) = (ks*NCG+cg)*256 + part*128 + L*4 + v4   (v = part*4+v4)
// Fragment for (ks,cg): lane L reads uint4 at base+L*4 and base+L*4+128.

#define XLD 520       // ushorts per x row (pad: conflict-free A-frag b128 loads)
#define HLD 72        // ushorts per h row

// Double-buffered weight regions; sX (startup only: A frags get cached in
// registers before any W2 issue) aliases the W2 buffers.
#define OFF_W2  0                                   // 2 x 65536  (sX needs 66560)
#define OFF_W1  131072                              // 2 x 65536  (64 KB per fc)
#define OFF_H   262144                              // 64*72*2 = 9216
#define OFF_PAD 271360                              // 64*4
#define SMEM_BYTES 271616                           // ~265 KB, 1 WG / WGP

#define W1_FC_U4 4096       // 64 KB per fc block (4 contiguous kp panels)
#define W2_FC_U4 4096       // 64 KB per fc block
#define WS_NEED ((size_t)(NFC*W1_FC_U4 + NFC*W2_FC_U4) * 16)   // 4 MB

#if __has_builtin(__builtin_amdgcn_global_load_async_to_lds_b128)
#define ASYNC_COPY 1
typedef __attribute__((address_space(1))) v4i* g4p_t;   // global int4*
typedef __attribute__((address_space(3))) v4i* l4p_t;   // LDS int4*
#else
#define ASYNC_COPY 0
#endif

// wait until <= N async ops outstanding (in-order completion => oldest finish)
template <int N>
__device__ __forceinline__ void asyncWaitLe() {
#if ASYNC_COPY
#if __has_builtin(__builtin_amdgcn_s_wait_asynccnt)
  __builtin_amdgcn_s_wait_asynccnt(N);
#else
  asm volatile("s_wait_asynccnt %0" :: "i"(N) : "memory");
#endif
#endif
}

#if ASYNC_COPY
// J*4096 is a template-parameter expression -> valid ICE for the builtin.
// ISA: INST_OFFSET is added to BOTH the LDS (VDST) and global (VADDR) addresses.
template <int J, int N>
__device__ __forceinline__ void asyncCopySteps(g4p_t s, l4p_t d) {
  if constexpr (J < N) {
    __builtin_amdgcn_global_load_async_to_lds_b128(s, d, J * 4096, 0);
    asyncCopySteps<J + 1, N>(s, d);
  }
}
#endif

// issue copy of N4 uint4 global -> LDS (no wait; 16 async ops for N4=4096)
template <int N4>
__device__ __forceinline__ void issueCopy(unsigned int* dst, const uint4* src, int tid) {
#if ASYNC_COPY
  asyncCopySteps<0, N4 / 256>((g4p_t)(src + tid), (l4p_t)(dst + 4 * tid));
#else
#pragma unroll
  for (int j = 0; j < N4 / 256; ++j)
    *(uint4*)(dst + 4 * (tid + j * 256)) = src[tid + j * 256];
#endif
}

// convert+swizzle W1 panel [128 x 64] (f32 row-major) -> B layout (bf16 K-pairs)
__device__ __forceinline__ void stageW1_convert(unsigned int* dst, const float* fw,
                                                int f0, int kp, int tid) {
  for (int i = tid; i < 1024; i += 256) {
    int k2 = i >> 4, c4 = i & 15;
    const float* g = fw + (long)(kp * KP + 2 * k2) * FDIM + f0 + c4 * 4;
    float4 e = *(const float4*)g;
    float4 o = *(const float4*)(g + FDIM);
    int ks = k2 >> 4, hf2 = (k2 >> 3) & 1, v = k2 & 7;
    int cg = c4 >> 2, n0 = (c4 & 3) * 4;
    unsigned int* d = dst + (ks * W1NCG + cg) * 256 + (v >> 2) * 128 + (hf2 * 16 + n0) * 4 + (v & 3);
    d[0] = cvt2bf(e.x, o.x); d[4]  = cvt2bf(e.y, o.y);
    d[8] = cvt2bf(e.z, o.z); d[12] = cvt2bf(e.w, o.w);
  }
}
// convert+swizzle W2 chunk [64 x 512] -> B layout
__device__ __forceinline__ void stageW2_convert(unsigned int* dst, const float* ow,
                                                int f0, int tid) {
  for (int i = tid; i < 4096; i += 256) {
    int k2 = i >> 7, c4 = i & 127;
    const float* g = ow + (long)(f0 + 2 * k2) * HDIM + c4 * 4;
    float4 e = *(const float4*)g;
    float4 o = *(const float4*)(g + HDIM);
    int ks = k2 >> 4, hf2 = (k2 >> 3) & 1, v = k2 & 7;
    int cg = c4 >> 2, n0 = (c4 & 3) * 4;
    unsigned int* d = dst + (ks * W2NCG + cg) * 256 + (v >> 2) * 128 + (hf2 * 16 + n0) * 4 + (v & 3);
    d[0] = cvt2bf(e.x, o.x); d[4]  = cvt2bf(e.y, o.y);
    d[8] = cvt2bf(e.z, o.z); d[12] = cvt2bf(e.w, o.w);
  }
}

// one-shot weight pre-convert into the LDS-image layout (d_ws)
__global__ __launch_bounds__(256, 1)
void convert_weights(const float* __restrict__ fw, const float* __restrict__ ow,
                     uint4* __restrict__ wsW1, uint4* __restrict__ wsW2) {
  extern __shared__ unsigned char csm[];
  unsigned int* sm = (unsigned int*)csm;
  const int b = blockIdx.x, tid = threadIdx.x;
  if (b < 128) {                                   // W1 region: fc=b>>2, kp=b&3
    stageW1_convert(sm, fw, (b >> 2) * FC, b & 3, tid);
    __syncthreads();
    uint4* dst = wsW1 + (size_t)b * 1024;
    for (int i = tid; i < 1024; i += 256) dst[i] = ((const uint4*)sm)[i];
  } else {                                         // W2 region: fc=b-128
    stageW2_convert(sm, ow, (b - 128) * FC, tid);
    __syncthreads();
    uint4* dst = wsW2 + (size_t)(b - 128) * W2_FC_U4;
    for (int i = tid; i < W2_FC_U4; i += 256) dst[i] = ((const uint4*)sm)[i];
  }
}

// A fragment (16x32 bf16) from row-major bf16 LDS: 2 x ds_load_b128
__device__ __forceinline__ void loadA(Frag& A, const unsigned short* base, int ld,
                                      int row0, int col0, int m, int hlf) {
  const unsigned int* p =
      (const unsigned int*)(base + (row0 + m) * ld) + (col0 >> 1) + (hlf << 3);
  A.q[0] = *(const uint4*)p;
  A.q[1] = *(const uint4*)(p + 4);
}
// B fragment: 2 x ds_load_b128 at +0 / +128 words
__device__ __forceinline__ void loadB(Frag& B, const unsigned int* fragBase) {
  B.q[0] = *(const uint4*)fragBase;
  B.q[1] = *(const uint4*)(fragBase + 128);
}

#define WMMA(a, b, c) __builtin_amdgcn_wmma_f32_16x16x32_bf16( \
    false, (a).v, false, (b).v, (short)0, (c), false, false)

template <bool PRE>
__global__ __launch_bounds__(256, 1)
void ffn_fused_wmma(const float* __restrict__ x, const float* __restrict__ padding,
                    const float* __restrict__ fw, const float* __restrict__ fb,
                    const float* __restrict__ ow, const float* __restrict__ ob,
                    const uint4* __restrict__ wsW1, const uint4* __restrict__ wsW2,
                    float* __restrict__ out) {
  extern __shared__ unsigned char smem[];
  unsigned short* sX  = (unsigned short*)(smem + OFF_W2);   // startup only
  unsigned int*   w2b = (unsigned int*)  (smem + OFF_W2);   // 2 x 16384 words
  unsigned int*   w1b = (unsigned int*)  (smem + OFF_W1);   // 2 x 16384 words
  unsigned short* sH  = (unsigned short*)(smem + OFF_H);
  float*          sPd = (float*)         (smem + OFF_PAD);

  const int tid  = threadIdx.x;
  const int wave = tid >> 5;
  const int lane = tid & 31;
  const int hlf  = lane >> 4;
  const int ln   = lane & 15;
  const long tokenBase = (long)blockIdx.x * TM;

  const int rA  = wave >> 1;            // phase-A: (row rA, cols cA0,cA0+1)
  const int cA0 = (wave & 1) * 2;
  const int rpB = (wave & 1) * 2;       // phase-B: rows rpB,rpB+1 x cgB..cgB+7
  const int cgB = (wave >> 1) * 8;

  // prefetch W1[0] (does not touch the sX-aliased region)
  if (PRE) issueCopy<W1_FC_U4>(w1b, wsW1, tid);

  // ---- stage x tile: 64 x 512 f32 -> bf16 ----
  for (int i = tid; i < TM * HDIM / 4; i += 256) {
    int row = i >> 7, c4 = i & 127;
    const float4 v = *(const float4*)(x + (tokenBase + row) * HDIM + c4 * 4);
    unsigned int* dst = (unsigned int*)(sX + row * XLD + c4 * 4);
    dst[0] = cvt2bf(v.x, v.y);
    dst[1] = cvt2bf(v.z, v.w);
  }
  if (tid < TM) sPd[tid] = padding[tokenBase + tid];
  __syncthreads();

  // ---- cache this wave's A fragments for ALL of K=512 in registers ----
  Frag Ax[16];
#pragma unroll
  for (int t = 0; t < 16; ++t)
    loadA(Ax[t], sX, XLD, rA * 16, t * 32, ln, hlf);
  __syncthreads();                       // all waves done reading sX
  if (PRE) issueCopy<W2_FC_U4>(w2b, wsW2, tid);   // W2[0] (overwrites sX region)

  v8f accO[16];
#pragma unroll
  for (int i = 0; i < 16; ++i) accO[i] = v8f{0,0,0,0,0,0,0,0};

  for (int fc = 0; fc < NFC; ++fc) {
    const int f0 = fc * FC;
    const int p = fc & 1;
    unsigned int* w1cur = w1b + p * 16384;
    unsigned int* w2cur = w2b + p * 16384;
    v8f accH0 = v8f{0,0,0,0,0,0,0,0};
    v8f accH1 = v8f{0,0,0,0,0,0,0,0};

    // ---- make W1[fc] ready; prefetch W1[fc+1] ----
    if (PRE) {
      asyncWaitLe<16>();        // oldest 16 = W1[fc] complete (W2[fc] may pend)
    } else {
      __syncthreads();
#pragma unroll
      for (int kp = 0; kp < 4; ++kp)
        stageW1_convert(w1cur + kp * 4096, fw, f0, kp, tid);
    }
    __syncthreads();
    if (PRE && fc + 1 < NFC)
      issueCopy<W1_FC_U4>(w1b + (p ^ 1) * 16384, wsW1 + (size_t)(fc + 1) * W1_FC_U4, tid);

    // ---------- phase A: h_chunk = x @ W1[:, f0:f0+64] ----------
    {
      const unsigned int* w1f = w1cur + lane * 4;
#pragma unroll
      for (int kp = 0; kp < 4; ++kp)
#pragma unroll
        for (int kk = 0; kk < 4; ++kk) {
          Frag B0; loadB(B0, w1f + kp * 4096 + (kk * W1NCG + cA0) * 256);
          Frag B1; loadB(B1, w1f + kp * 4096 + (kk * W1NCG + cA0 + 1) * 256);
          accH0 = WMMA(Ax[kp * 4 + kk], B0, accH0);
          accH1 = WMMA(Ax[kp * 4 + kk], B1, accH1);
        }
    }
    // write h chunk (bias add, bf16) to sH
    {
      float b1a = fb[f0 + cA0 * 16 + ln];
      float b1b = fb[f0 + (cA0 + 1) * 16 + ln];
#pragma unroll
      for (int v = 0; v < 8; ++v) {
        int m = rA * 16 + v + 8 * hlf;
        sH[m * HLD + cA0 * 16 + ln]       = f2bf(accH0[v] + b1a);
        sH[m * HLD + (cA0 + 1) * 16 + ln] = f2bf(accH1[v] + b1b);
      }
    }

    // ---- make W2[fc] ready; prefetch W2[fc+1] ----
    if (PRE) {
      if (fc + 1 < NFC) asyncWaitLe<16>();  // leaves W1[fc+1] outstanding
      else              asyncWaitLe<0>();   // last chunk: nothing newer pending
    } else {
      stageW2_convert(w2cur, ow, f0, tid);
    }
    __syncthreads();                        // W2[fc] + sH visible to all
    if (PRE && fc + 1 < NFC)
      issueCopy<W2_FC_U4>(w2b + (p ^ 1) * 16384, wsW2 + (size_t)(fc + 1) * W2_FC_U4, tid);

    // ---------- phase B: out += h_chunk @ W2[f0:f0+64, :] ----------
    {
      Frag A00, A10, A01, A11;
      loadA(A00, sH, HLD, (rpB + 0) * 16, 0,  ln, hlf);
      loadA(A10, sH, HLD, (rpB + 1) * 16, 0,  ln, hlf);
      loadA(A01, sH, HLD, (rpB + 0) * 16, 32, ln, hlf);
      loadA(A11, sH, HLD, (rpB + 1) * 16, 32, ln, hlf);
      const unsigned int* w2f = w2cur + lane * 4;
      Frag B0, B1;
      loadB(B0, w2f + cgB * 256);                         // kk=0, first cg
#pragma unroll
      for (int i = 0; i < 8; ++i) {
        int cg = cgB + i;
        loadB(B1, w2f + (W2NCG + cg) * 256);              // kk=1, this cg
        accO[i * 2 + 0] = WMMA(A00, B0, accO[i * 2 + 0]);
        accO[i * 2 + 1] = WMMA(A10, B0, accO[i * 2 + 1]);
        if (i < 7) loadB(B0, w2f + (cg + 1) * 256);       // kk=0, next cg
        accO[i * 2 + 0] = WMMA(A01, B1, accO[i * 2 + 0]);
        accO[i * 2 + 1] = WMMA(A11, B1, accO[i * 2 + 1]);
      }
    }
  }

  // ---------- epilogue: bias2 + padding mask + store ----------
#pragma unroll
  for (int i = 0; i < 8; ++i) {
    int col = (cgB + i) * 16 + ln;
    float b2 = ob[col];
#pragma unroll
    for (int rr = 0; rr < 2; ++rr) {
#pragma unroll
      for (int v = 0; v < 8; ++v) {
        int m = (rpB + rr) * 16 + v + 8 * hlf;
        float val = accO[i * 2 + rr][v] + b2;
        if (sPd[m] >= 1e-9f) val = 0.0f;   // padded token -> zero
        out[(tokenBase + m) * HDIM + col] = val;
      }
    }
  }
}

extern "C" void kernel_launch(void* const* d_in, const int* in_sizes, int n_in,
                              void* d_out, int out_size, void* d_ws, size_t ws_size,
                              hipStream_t stream) {
  (void)n_in; (void)out_size;
  const float* x  = (const float*)d_in[0];
  const float* pd = (const float*)d_in[1];
  const float* fw = (const float*)d_in[2];
  const float* fb = (const float*)d_in[3];
  const float* ow = (const float*)d_in[4];
  const float* ob = (const float*)d_in[5];
  float* out = (float*)d_out;

  const int M = in_sizes[1];           // B*L tokens (32768)
  const int blocks = M / TM;           // 512

  uint4* wsW1 = (uint4*)d_ws;
  uint4* wsW2 = wsW1 + NFC * W1_FC_U4;
  const bool pre = (d_ws != nullptr) && (ws_size >= WS_NEED);

  if (pre) {
    (void)hipFuncSetAttribute((const void*)convert_weights,
                              hipFuncAttributeMaxDynamicSharedMemorySize, 65536);
    convert_weights<<<160, 256, 65536, stream>>>(fw, ow, wsW1, wsW2);
    (void)hipFuncSetAttribute((const void*)ffn_fused_wmma<true>,
                              hipFuncAttributeMaxDynamicSharedMemorySize, SMEM_BYTES);
    ffn_fused_wmma<true><<<blocks, 256, SMEM_BYTES, stream>>>(
        x, pd, fw, fb, ow, ob, wsW1, wsW2, out);
  } else {
    (void)hipFuncSetAttribute((const void*)ffn_fused_wmma<false>,
                              hipFuncAttributeMaxDynamicSharedMemorySize, SMEM_BYTES);
    ffn_fused_wmma<false><<<blocks, 256, SMEM_BYTES, stream>>>(
        x, pd, fw, fb, ow, ob, wsW1, wsW2, out);
  }
}
